// PromptDetectionLoss_26800595927826
// MI455X (gfx1250) — compile-verified, matrix operand order checked by reading, not run on hardware
//
#include <hip/hip_runtime.h>
#include <hip/hip_bf16.h>
#include <math.h>

// ---- constants from the reference ----
#define TOPK 13
#define NEG_INF_F (-1e30f)
#define EPS_F 1e-7f
#define FLT_MAX_F 3.402823466e38f
#define MAXM 128
#define MAXC 256
#define NQ 8   // quantity columns (padded into a 16-wide WMMA tile logically)

__device__ __forceinline__ float sigm(float x) { return 1.0f / (1.0f + expf(-x)); }
__device__ __forceinline__ int clampi(int v, int lo, int hi) { return v < lo ? lo : (v > hi ? hi : v); }

// ---------------------------------------------------------------------------
// Kernel 0: zero the per-batch accumulators
// ---------------------------------------------------------------------------
__global__ void k_init(float* acc, int n) {
    int i = blockIdx.x * blockDim.x + threadIdx.x;
    if (i < n) acc[i] = 0.0f;
}

// ---------------------------------------------------------------------------
// Kernel 1: per-(b,m) top-13 threshold of align = sigmoid(cls) * iou^6
// One 256-thread block per GT row. Per-thread top-13 insertion sort, then 13
// rounds of parallel max-extraction over the 256*13 candidates in LDS.
// ---------------------------------------------------------------------------
__global__ void __launch_bounds__(256)
k_thresh(const float* __restrict__ ps, const float4* __restrict__ pb,
         const float* __restrict__ anc, const float4* __restrict__ gtb,
         const int* __restrict__ gtl, const unsigned char* __restrict__ cm,
         float* __restrict__ thr, int N, int C, int M)
{
    const int m = blockIdx.x, b = blockIdx.y, tid = threadIdx.x;
    __shared__ float s_vals[256 * TOPK];
    __shared__ float s_red[256];
    __shared__ int   s_redi[256];
    __shared__ float s_last;

    float4 g = gtb[(size_t)b * M + m];
    int lraw = gtl[(size_t)b * M + m];
    int lab = clampi(lraw, 0, C - 1);
    bool valid = (lraw >= 0) && (lraw < C) && (cm[(size_t)b * C + lab] != 0);
    float ga = (g.z - g.x) * (g.w - g.y);

    float top[TOPK];
#pragma unroll
    for (int i = 0; i < TOPK; i++) top[i] = -FLT_MAX_F;

    for (int n = tid; n < N; n += blockDim.x) {
        float ax = anc[2 * n], ay = anc[2 * n + 1];
        float4 p = pb[(size_t)b * N + n];
        float x1 = fmaxf(p.x, g.x), y1 = fmaxf(p.y, g.y);
        float x2 = fminf(p.z, g.z), y2 = fminf(p.w, g.w);
        float inter = fmaxf(x2 - x1, 0.f) * fmaxf(y2 - y1, 0.f);
        float pa = (p.z - p.x) * (p.w - p.y);
        float iou = inter / (pa + ga - inter + EPS_F);
        bool inside = (ax >= g.x) && (ax <= g.z) && (ay >= g.y) && (ay <= g.w);
        float align = NEG_INF_F;
        if (inside && valid) {
            float cls = sigm(ps[((size_t)b * N + n) * C + lab]); // L2-resident gather
            float i2 = iou * iou;
            align = cls * (i2 * i2 * i2);                         // alpha=1, beta=6
        }
        if (align > top[TOPK - 1]) {
            int i = TOPK - 1;
            while (i > 0 && top[i - 1] < align) { top[i] = top[i - 1]; --i; }
            top[i] = align;
        }
    }
#pragma unroll
    for (int i = 0; i < TOPK; i++) s_vals[tid * TOPK + i] = top[i];
    __syncthreads();

    for (int round = 0; round < TOPK; ++round) {
        float mv = -FLT_MAX_F; int mi = tid * TOPK;
        for (int i = 0; i < TOPK; i++) {
            float v = s_vals[tid * TOPK + i];
            if (v > mv) { mv = v; mi = tid * TOPK + i; }
        }
        s_red[tid] = mv; s_redi[tid] = mi;
        __syncthreads();
        for (int s = 128; s > 0; s >>= 1) {
            if (tid < s && s_red[tid + s] > s_red[tid]) {
                s_red[tid] = s_red[tid + s]; s_redi[tid] = s_redi[tid + s];
            }
            __syncthreads();
        }
        if (tid == 0) { s_last = s_red[0]; s_vals[s_redi[0]] = -FLT_MAX_F; }
        __syncthreads();
    }
    if (tid == 0) thr[(size_t)b * M + m] = s_last; // 13th-largest value
}

// ---------------------------------------------------------------------------
// Kernel 2: per-anchor assignment — argmax over GTs of candidate aligns
// ---------------------------------------------------------------------------
__global__ void __launch_bounds__(256)
k_assign(const float* __restrict__ ps, const float4* __restrict__ pb,
         const float* __restrict__ anc, const float4* __restrict__ gtb,
         const int* __restrict__ gtl, const unsigned char* __restrict__ cm,
         const float* __restrict__ thr, int* __restrict__ fg, int* __restrict__ mg,
         float* __restrict__ ov, int N, int C, int M)
{
    const int b = blockIdx.y;
    const int n = blockIdx.x * blockDim.x + threadIdx.x;
    __shared__ float4        s_g[MAXM];
    __shared__ int           s_lab[MAXM];
    __shared__ unsigned char s_val[MAXM];
    __shared__ float         s_thr[MAXM];
    for (int i = threadIdx.x; i < M; i += blockDim.x) {
        s_g[i] = gtb[(size_t)b * M + i];
        int lr = gtl[(size_t)b * M + i];
        int lab = clampi(lr, 0, C - 1);
        s_lab[i] = lab;
        s_val[i] = (lr >= 0 && lr < C && cm[(size_t)b * C + lab] != 0) ? 1 : 0;
        s_thr[i] = thr[(size_t)b * M + i];
    }
    __syncthreads();
    if (n >= N) return;

    float ax = anc[2 * n], ay = anc[2 * n + 1];
    float4 p = pb[(size_t)b * N + n];
    float pa = (p.z - p.x) * (p.w - p.y);
    const float* psrow = ps + ((size_t)b * N + n) * C;

    float best = -FLT_MAX_F; int bm = 0; float biou = 0.f;
    for (int m = 0; m < M; m++) {
        float4 g = s_g[m];
        float x1 = fmaxf(p.x, g.x), y1 = fmaxf(p.y, g.y);
        float x2 = fminf(p.z, g.z), y2 = fminf(p.w, g.w);
        float inter = fmaxf(x2 - x1, 0.f) * fmaxf(y2 - y1, 0.f);
        float ga = (g.z - g.x) * (g.w - g.y);
        float iou = inter / (pa + ga - inter + EPS_F);
        float align = NEG_INF_F;
        if (s_val[m] && ax >= g.x && ax <= g.z && ay >= g.y && ay <= g.w) {
            float cls = sigm(psrow[s_lab[m]]);
            float i2 = iou * iou;
            align = cls * (i2 * i2 * i2);
        }
        float cand = (align >= s_thr[m]) ? align : NEG_INF_F; // top-13 membership
        if (cand > best) { best = cand; bm = m; biou = iou; } // first-max argmax
    }
    int isfg = (best > -1.0f) ? 1 : 0;
    fg[(size_t)b * N + n] = isfg;
    mg[(size_t)b * N + n] = bm;
    ov[(size_t)b * N + n] = biou;
}

// ---------------------------------------------------------------------------
// Kernel 3: fused per-anchor losses + WMMA block reduction + atomic accumulate
// acc[b][0..7] = {bce_sum, (1-ciou)_sum, dfl_sum, cnt, pos_sum, neg_sum, miou_sum, 0}
// ---------------------------------------------------------------------------
__global__ void __launch_bounds__(256)
k_main(const float* __restrict__ ps, const float4* __restrict__ pb,
       const float* __restrict__ anc, const float* __restrict__ strt,
       const float* __restrict__ bd, const float4* __restrict__ gtb,
       const int* __restrict__ gtl, const unsigned char* __restrict__ cm,
       const int* __restrict__ fg, const int* __restrict__ mg,
       const float* __restrict__ ov, float* __restrict__ acc,
       int N, int C, int M, int R)
{
    const int b = blockIdx.y;
    const int tid = threadIdx.x;
    const int n = blockIdx.x * blockDim.x + tid;
    __shared__ float4        s_g[MAXM];
    __shared__ int           s_lab[MAXM];
    __shared__ unsigned char s_cm[MAXC];
    // Column-major partials: s_part[q][row], row = tid. Column-major makes the
    // WMMA B-operand fetch contiguous (rows r, r+1 -> one ds_load_b64).
    __shared__ float s_part[NQ * 256];
    for (int i = tid; i < M; i += blockDim.x) {
        s_g[i] = gtb[(size_t)b * M + i];
        s_lab[i] = clampi(gtl[(size_t)b * M + i], 0, C - 1);
    }
    for (int i = tid; i < C; i += blockDim.x) s_cm[i] = cm[(size_t)b * C + i];
    __syncthreads();

    float q0 = 0.f, q1 = 0.f, q2 = 0.f, q3 = 0.f, q4 = 0.f, q5 = 0.f, q6 = 0.f;
    if (n < N) {
        int   isfg  = fg[(size_t)b * N + n];
        int   m     = mg[(size_t)b * N + n];
        float oviou = ov[(size_t)b * N + n];
        float4 p = pb[(size_t)b * N + n];
        float4 t = isfg ? s_g[m] : p;
        int   mlab = s_lab[m];
        float tsc = fmaxf(oviou, 0.1f);

        // BCE over C classes (second pass over ps is an L2 hit: 80MB < 192MB L2)
        const float* psrow = ps + ((size_t)b * N + n) * C;
        float bces = 0.f, mx = NEG_INF_F, posp = 0.f;
        for (int c = 0; c < C; c++) {
            float s  = psrow[c];
            float vc = s_cm[c] ? 1.f : 0.f;
            float tgt = (isfg && c == mlab) ? tsc : 0.f;
            float bce = fmaxf(s, 0.f) - s * tgt + log1pf(expf(-fabsf(s)));
            bces += bce * vc;
            float pr = 1.f / (1.f + expf(-s));
            if (vc > 0.f) mx = fmaxf(mx, pr);
            if (c == mlab) posp = pr;
        }
        q0 = bces;

        // CIoU(pb, tb)
        float x1 = fmaxf(p.x, t.x), y1 = fmaxf(p.y, t.y);
        float x2 = fminf(p.z, t.z), y2 = fminf(p.w, t.w);
        float inter = fmaxf(x2 - x1, 0.f) * fmaxf(y2 - y1, 0.f);
        float a1 = (p.z - p.x) * (p.w - p.y), a2 = (t.z - t.x) * (t.w - t.y);
        float iou = inter / (a1 + a2 - inter + EPS_F);
        float w1 = p.z - p.x, h1 = p.w - p.y, w2 = t.z - t.x, h2 = t.w - t.y;
        float cwv = fmaxf(p.z, t.z) - fminf(p.x, t.x);
        float chv = fmaxf(p.w, t.w) - fminf(p.y, t.y);
        float c2 = cwv * cwv + chv * chv + EPS_F;
        float dx = t.x + t.z - p.x - p.z, dy = t.y + t.w - p.y - p.w;
        float rho2 = (dx * dx + dy * dy) * 0.25f;
        float dat = atanf(w2 / (h2 + EPS_F)) - atanf(w1 / (h1 + EPS_F));
        float v = 0.405284734569351f * dat * dat;              // (4/pi^2) * (...)^2
        float alpha = v / (v - iou + (1.f + EPS_F));
        float ciou = iou - (rho2 / c2 + v * alpha);

        if (isfg) {
            q1 = 1.f - ciou;
            q3 = 1.f;
            q4 = posp;
            q6 = iou;
            // DFL (only fg contributes)
            float ax = anc[2 * n], ay = anc[2 * n + 1];
            float st = strt[n];
            float dv[4] = { (ax - t.x) / st, (ay - t.y) / st, (t.z - ax) / st, (t.w - ay) / st };
            float dfl = 0.f;
            for (int k = 0; k < 4; k++) {
                float d = fminf(fmaxf(dv[k], 0.f), (float)(R - 1) - 0.01f);
                int tl = (int)d;
                int tr = tl + 1 < R - 1 ? tl + 1 : R - 1;
                float wl = (float)tr - d, wr = 1.f - wl;
                const float* row = bd + (((size_t)b * N + n) * 4 + k) * R;
                float mv = -FLT_MAX_F;
                for (int i = 0; i < R; i++) mv = fmaxf(mv, row[i]);
                float se = 0.f;
                for (int i = 0; i < R; i++) se += expf(row[i] - mv);
                float lse = mv + logf(se);
                dfl += (lse - row[tl]) * wl + (lse - row[tr]) * wr;
            }
            q2 = dfl;
        } else {
            q5 = mx;
        }
    }
    s_part[0 * 256 + tid] = q0; s_part[1 * 256 + tid] = q1;
    s_part[2 * 256 + tid] = q2; s_part[3 * 256 + tid] = q3;
    s_part[4 * 256 + tid] = q4; s_part[5 * 256 + tid] = q5;
    s_part[6 * 256 + tid] = q6; s_part[7 * 256 + tid] = 0.f;
    __syncthreads();

    // ---- WMMA column-sum reduction over the 256x8 partial matrix ----
    // D = ones(16x4) x B(4x16) accumulated over 64 K-chunks:
    //   D[i][j] = sum_k B[k][j]  (every D row identical = column sums).
    // Key facts exploited:
    //  * D column j depends only on B column j, and we read only columns 0..6,
    //    so lanes holding columns 8..15 just mirror column (col&7) -> no
    //    conditionals, no EXEC masking in the loop.
    //  * Any permutation of K-rows leaves a column sum unchanged, so we map
    //    the two B registers of each lane to adjacent rows (r, r+1), making
    //    the fetch a single contiguous 8-byte LDS read (ds_load_b64).
    // Wave 0 only; branch is wave-uniform so EXEC is all-ones for the WMMAs.
    if (tid < 32) {
        const int l = tid, col = l & 15, half = l >> 4;
        const int cc = col & 7;              // mirrored column for lanes 8..15
        const float* colbase = &s_part[cc * 256];
#if defined(__HIP_DEVICE_COMPILE__) && __has_builtin(__builtin_amdgcn_wmma_f32_16x16x4_f32)
        typedef __attribute__((ext_vector_type(2))) float v2f;
        typedef __attribute__((ext_vector_type(8))) float v8f;
        v2f a; a[0] = 1.0f; a[1] = 1.0f;
        v8f c = {};
        for (int it = 0; it < 256 / 4; ++it) {
            const int r = it * 4 + 2 * half; // halves cover rows {0,1} and {2,3}
            v2f bm;
            bm[0] = colbase[r];
            bm[1] = colbase[r + 1];
            c = __builtin_amdgcn_wmma_f32_16x16x4_f32(false, a, false, bm,
                                                      (short)0, c, false, false);
        }
        if (l < 7) atomicAdd(&acc[(size_t)b * NQ + l], c[0]);
#elif defined(__HIP_DEVICE_COMPILE__)
        // Fallback: codegen-confirmed f16 WMMA (same ones-matrix column-sum trick)
        typedef __attribute__((ext_vector_type(16))) _Float16 v16h;
        typedef __attribute__((ext_vector_type(8))) float v8f;
        v16h a;
        for (int i = 0; i < 16; i++) a[i] = (_Float16)1.0f;
        v8f c = {};
        for (int it = 0; it < 256 / 32; ++it) {
            v16h bm;
            for (int i = 0; i < 16; i++) bm[i] = (_Float16)colbase[it * 32 + half * 16 + i];
            c = __builtin_amdgcn_wmma_f32_16x16x32_f16(false, a, false, bm,
                                                       (short)0, c, false, false);
        }
        if (l < 7) atomicAdd(&acc[(size_t)b * NQ + l], c[0]);
#else
        // Host-pass parse-only path (never executed on device)
        if (l == 0) {
            for (int q = 0; q < 7; q++) {
                float s = 0.f;
                for (int r = 0; r < 256; r++) s += s_part[q * 256 + r];
                atomicAdd(&acc[(size_t)b * NQ + q], s);
            }
        }
#endif
    }
}

// ---------------------------------------------------------------------------
// Kernel 4: per-sample normalization + batch combine -> 8 scalar outputs
// ---------------------------------------------------------------------------
__global__ void k_finalize(const float* __restrict__ acc,
                           const unsigned char* __restrict__ cm,
                           float* __restrict__ out, int B, int N, int C)
{
    __shared__ float sv[64][8];
    int b = threadIdx.x;
    if (b < B && b < 64) {
        float vcs = 0.f;
        for (int c = 0; c < C; c++) vcs += cm[(size_t)b * C + c] ? 1.f : 0.f;
        float matchr = acc[(size_t)b * NQ + 0];
        float cious  = acc[(size_t)b * NQ + 1];
        float dflr   = acc[(size_t)b * NQ + 2];
        float cnt    = acc[(size_t)b * NQ + 3];
        float pos    = acc[(size_t)b * NQ + 4];
        float neg    = acc[(size_t)b * NQ + 5];
        float miou   = acc[(size_t)b * NQ + 6];
        sv[b][0] = matchr / ((float)N * fmaxf(vcs, 1.f));
        sv[b][1] = (cnt > 0.f) ? cious / fmaxf(cnt, 1.f) : 0.f;
        sv[b][2] = (cnt > 0.f) ? dflr / fmaxf(cnt * 4.f, 1.f) : 0.f;
        sv[b][3] = cnt;
        sv[b][4] = (float)N - cnt;
        sv[b][5] = pos;
        sv[b][6] = neg;
        sv[b][7] = miou;
    }
    __syncthreads();
    if (threadIdx.x == 0) {
        float t[8] = {0, 0, 0, 0, 0, 0, 0, 0};
        for (int i = 0; i < B; i++)
            for (int q = 0; q < 8; q++) t[q] += sv[i][q];
        float invB = 1.f / (float)B;
        out[0] = (0.5f * t[0] + 7.5f * t[1] + 1.5f * t[2]) * invB; // loss
        out[1] = t[0] * invB;                                      // total_match/B
        out[2] = t[1] * invB;                                      // total_iou/B
        out[3] = t[2] * invB;                                      // total_dfl/B
        out[4] = t[3];                                             // tot_pos
        out[5] = t[5] / fmaxf(t[3], 1.f);                          // mean_pos
        out[6] = t[6] / fmaxf(t[4], 1.f);                          // mean_neg
        out[7] = t[7] / fmaxf(t[3], 1.f);                          // mean_miou
    }
}

// ---------------------------------------------------------------------------
extern "C" void kernel_launch(void* const* d_in, const int* in_sizes, int n_in,
                              void* d_out, int out_size, void* d_ws, size_t ws_size,
                              hipStream_t stream)
{
    const float*  ps  = (const float*)d_in[0];          // (B,N,C)
    const float4* pbx = (const float4*)d_in[1];         // (B,N,4)
    const float*  anc = (const float*)d_in[2];          // (N,2)
    const float*  st  = (const float*)d_in[3];          // (N,1)
    const float*  bd  = (const float*)d_in[4];          // (B,N,4R)
    const float4* gtb = (const float4*)d_in[5];         // (B,M,4)
    const int*    gtl = (const int*)d_in[6];            // (B,M)
    const unsigned char* cm = (const unsigned char*)d_in[7]; // (B,C) bool

    const int N = in_sizes[2] / 2;
    const int B = in_sizes[1] / (4 * N);
    const int C = in_sizes[7] / B;
    const int M = in_sizes[6] / B;
    const int R = in_sizes[4] / (B * N * 4);

    // workspace layout (all 4-byte aligned)
    float* thr = (float*)d_ws;                       // B*M
    int*   fg  = (int*)(thr + (size_t)B * M);        // B*N
    int*   mg  = fg + (size_t)B * N;                 // B*N
    float* ov  = (float*)(mg + (size_t)B * N);       // B*N
    float* acc = ov + (size_t)B * N;                 // B*NQ
    float* out = (float*)d_out;

    k_init<<<(B * NQ + 255) / 256, 256, 0, stream>>>(acc, B * NQ);
    k_thresh<<<dim3(M, B), 256, 0, stream>>>(ps, pbx, anc, gtb, gtl, cm, thr, N, C, M);
    const int nb = (N + 255) / 256;
    k_assign<<<dim3(nb, B), 256, 0, stream>>>(ps, pbx, anc, gtb, gtl, cm, thr,
                                              fg, mg, ov, N, C, M);
    k_main<<<dim3(nb, B), 256, 0, stream>>>(ps, pbx, anc, st, bd, gtb, gtl, cm,
                                            fg, mg, ov, acc, N, C, M, R);
    k_finalize<<<1, 64, 0, stream>>>(acc, cm, out, B, N, C);
}
